// RWKV7Model_71485435675231
// MI455X (gfx1250) — matrix operand order repaired and checked by reading, not compiled, not run on hardware
//
#include <hip/hip_runtime.h>

// ---------------- model constants ----------------
#define BB    2
#define TT    2048
#define CC    1024
#define HSZ   64
#define NH    16
#define LR    8
#define VOC   32000
#define FFD   4096
#define BT    (BB*TT)

typedef __bf16 bf16;
typedef __attribute__((ext_vector_type(8)))  float    v8f;
typedef __attribute__((ext_vector_type(16))) bf16     v16bf;
typedef __attribute__((ext_vector_type(4)))  unsigned v4u;
typedef __attribute__((ext_vector_type(8)))  unsigned v8u;

// =================================================================
// elementwise fp32 -> bf16 conversion (weights, once per launch)
// =================================================================
__global__ void k_cvt(const float* __restrict__ in, bf16* __restrict__ out, size_t n) {
    size_t i = ((size_t)blockIdx.x * blockDim.x + threadIdx.x) * 4;
    if (i + 3 < n) {
        float4 v = *(const float4*)(in + i);
        out[i+0] = (bf16)v.x; out[i+1] = (bf16)v.y;
        out[i+2] = (bf16)v.z; out[i+3] = (bf16)v.w;
    } else {
        for (size_t j = i; j < n; j++) out[j] = (bf16)in[j];
    }
}

// =================================================================
// embedding gather: x[row,:] = emb[token[row],:]
// =================================================================
__global__ __launch_bounds__(256) void k_embed(const int* __restrict__ tok,
                                               const float* __restrict__ emb,
                                               float* __restrict__ x) {
    int row = blockIdx.x;
    int t = tok[row];
    const float4* src = (const float4*)(emb + (size_t)t * CC);
    float4* dst = (float4*)(x + (size_t)row * CC);
    dst[threadIdx.x] = src[threadIdx.x];
}

// =================================================================
// LayerNorm over C=1024, one block (256 thr, 4 elems/thr) per row.
// =================================================================
__global__ __launch_bounds__(256) void k_layernorm(const float* __restrict__ in,
                                                   const float* __restrict__ w,
                                                   const float* __restrict__ b,
                                                   float* __restrict__ out,
                                                   bf16* __restrict__ outb) {
    __shared__ float s1[256], s2[256];
    int row = blockIdx.x, tid = threadIdx.x;
    const float* p = in + (size_t)row * CC;
    float4 v = ((const float4*)p)[tid];
    s1[tid] = v.x + v.y + v.z + v.w;
    s2[tid] = v.x*v.x + v.y*v.y + v.z*v.z + v.w*v.w;
    __syncthreads();
    for (int st = 128; st > 0; st >>= 1) {
        if (tid < st) { s1[tid] += s1[tid+st]; s2[tid] += s2[tid+st]; }
        __syncthreads();
    }
    float mean = s1[0] * (1.0f / CC);
    float var  = s2[0] * (1.0f / CC) - mean * mean;
    float rs   = rsqrtf(var + 1e-5f);
    int c = tid * 4;
    float4 o;
    o.x = (v.x - mean) * rs * w[c+0] + b[c+0];
    o.y = (v.y - mean) * rs * w[c+1] + b[c+1];
    o.z = (v.z - mean) * rs * w[c+2] + b[c+2];
    o.w = (v.w - mean) * rs * w[c+3] + b[c+3];
    ((float4*)(out + (size_t)row * CC))[tid] = o;
    if (outb) {
        size_t base = (size_t)row * CC + c;
        outb[base+0] = (bf16)o.x; outb[base+1] = (bf16)o.y;
        outb[base+2] = (bf16)o.z; outb[base+3] = (bf16)o.w;
    }
}

// =================================================================
// tmix token-shift mixing: xx = shift(xn)-xn ; out_j = xn + xx*coef_j
// =================================================================
__global__ __launch_bounds__(256) void k_tmix_mix(const float* __restrict__ xn,
        const float* __restrict__ cr, const float* __restrict__ cw,
        const float* __restrict__ ck, const float* __restrict__ cv,
        const float* __restrict__ ca, const float* __restrict__ cg,
        bf16* __restrict__ oxr, bf16* __restrict__ oxw, bf16* __restrict__ oxk,
        bf16* __restrict__ oxv, bf16* __restrict__ oxa, bf16* __restrict__ oxg) {
    size_t gid = (size_t)blockIdx.x * blockDim.x + threadIdx.x;
    int c = (int)(gid & (CC - 1));
    size_t row = gid >> 10;
    int t = (int)(row & (TT - 1));
    float xv = xn[gid];
    float px = (t == 0) ? 0.0f : xn[gid - CC];
    float xx = px - xv;
    oxr[gid] = (bf16)(xv + xx * cr[c]);
    oxw[gid] = (bf16)(xv + xx * cw[c]);
    oxk[gid] = (bf16)(xv + xx * ck[c]);
    oxv[gid] = (bf16)(xv + xx * cv[c]);
    oxa[gid] = (bf16)(xv + xx * ca[c]);
    oxg[gid] = (bf16)(xv + xx * cg[c]);
}

__global__ __launch_bounds__(256) void k_ffn_mix(const float* __restrict__ xn,
                                                 const float* __restrict__ ck,
                                                 bf16* __restrict__ o) {
    size_t gid = (size_t)blockIdx.x * blockDim.x + threadIdx.x;
    int c = (int)(gid & (CC - 1));
    size_t row = gid >> 10;
    int t = (int)(row & (TT - 1));
    float xv = xn[gid];
    float px = (t == 0) ? 0.0f : xn[gid - CC];
    o[gid] = (bf16)(xv + (px - xv) * ck[c]);
}

// =================================================================
// LoRA stage1: (BT,C) @ (C,8) -> (BT,8), act: 0=id 1=tanh 2=sigmoid
// =================================================================
__global__ __launch_bounds__(256) void k_lora_s1(const bf16* __restrict__ X,
                                                 const float* __restrict__ W1,
                                                 float* __restrict__ out, int act) {
    __shared__ float red[256 * LR];
    int row = blockIdx.x, tid = threadIdx.x;
    float acc[LR];
#pragma unroll
    for (int j = 0; j < LR; j++) acc[j] = 0.0f;
    for (int k = tid; k < CC; k += 256) {
        float xv = (float)X[(size_t)row * CC + k];
        const float* wp = W1 + (size_t)k * LR;
#pragma unroll
        for (int j = 0; j < LR; j++) acc[j] += xv * wp[j];
    }
#pragma unroll
    for (int j = 0; j < LR; j++) red[tid * LR + j] = acc[j];
    __syncthreads();
    for (int st = 128; st > 0; st >>= 1) {
        if (tid < st)
            for (int j = 0; j < LR; j++) red[tid * LR + j] += red[(tid + st) * LR + j];
        __syncthreads();
    }
    if (tid < LR) {
        float v = red[tid];
        if (act == 1) v = tanhf(v);
        else if (act == 2) v = 1.0f / (1.0f + expf(-v));
        out[(size_t)row * LR + tid] = v;
    }
}

// =================================================================
// LoRA stage2: (BT,8) @ (8,C) + b0 then mode:
//   0: identity   1: sigmoid   2: decay = exp(-exp(-softplus(-t)-0.5))
// =================================================================
__global__ __launch_bounds__(256) void k_lora_s2(const float* __restrict__ h,
                                                 const float* __restrict__ W2,
                                                 const float* __restrict__ b0,
                                                 float* __restrict__ out, int mode) {
    size_t gid = (size_t)blockIdx.x * blockDim.x + threadIdx.x;
    int c = (int)(gid & (CC - 1));
    size_t row = gid >> 10;
    const float* hp = h + row * LR;
    float t = b0 ? b0[c] : 0.0f;
#pragma unroll
    for (int j = 0; j < LR; j++) t += hp[j] * W2[(size_t)j * CC + c];
    float o;
    if (mode == 1) {
        o = 1.0f / (1.0f + expf(-t));
    } else if (mode == 2) {
        float z = -t;   // softplus(-t), numerically stable
        float sp = (z > 0.0f) ? z + log1pf(expf(-z)) : log1pf(expf(z));
        float wv = -sp - 0.5f;
        o = expf(-expf(wv));
    } else {
        o = t;
    }
    out[gid] = o;
}

// v = v + (v_first - v) * gate
__global__ __launch_bounds__(256) void k_vmix(float* __restrict__ v,
                                              const float* __restrict__ vf,
                                              const float* __restrict__ gate) {
    size_t gid = (size_t)blockIdx.x * blockDim.x + threadIdx.x;
    float vv = v[gid];
    v[gid] = vv + (vf[gid] - vv) * gate[gid];
}

// =================================================================
// kk normalize + k update + a/b vectors; one 64-thread block per (b,t,h)
// =================================================================
__global__ __launch_bounds__(64) void k_kv_prep(const float* __restrict__ k,
        const float* __restrict__ aa, const float* __restrict__ kkp,
        const float* __restrict__ kap, float* __restrict__ k2,
        float* __restrict__ a_in, float* __restrict__ b_in) {
    __shared__ float s[64];
    int blk = blockIdx.x, i = threadIdx.x;
    size_t base = (size_t)blk * HSZ;
    int c = (blk % NH) * HSZ + i;
    float kv = k[base + i];
    float kk = kv * kkp[c];
    s[i] = kk * kk;
    __syncthreads();
    for (int st = 32; st > 0; st >>= 1) { if (i < st) s[i] += s[i + st]; __syncthreads(); }
    float inv = 1.0f / fmaxf(sqrtf(s[0]), 1e-12f);
    float kkn = kk * inv;
    float av = aa[base + i];
    a_in[base + i] = -kkn;
    b_in[base + i] = kkn * av;
    k2[base + i] = kv * (1.0f + (av - 1.0f) * kap[c]);
}

// =================================================================
// RWKV-7 sequential scan. One block per (b,h); 64x64 fp32 state held
// in registers (16 floats/thread). wave32 shfl_xor for 4-lane reduce.
// =================================================================
__global__ __launch_bounds__(256) void k_rwkv_scan(const float* __restrict__ r,
        const float* __restrict__ dec, const float* __restrict__ k,
        const float* __restrict__ v, const float* __restrict__ a,
        const float* __restrict__ b, float* __restrict__ y) {
    const int bh = blockIdx.x;               // b*NH + h
    const int bb = bh / NH, hh = bh % NH;
    const int tid = threadIdx.x;
    const int vi = tid >> 2;                 // 0..63  (state row)
    const int kg = tid & 3;                  // 0..3   (16-wide k group)
    const int kb = kg * 16;
    __shared__ float sr[HSZ], sw[HSZ], sk[HSZ], sv[HSZ], sa[HSZ], sb[HSZ];
    float S[16];
#pragma unroll
    for (int j = 0; j < 16; j++) S[j] = 0.0f;
    const size_t base = (size_t)bb * TT * CC + (size_t)hh * HSZ;
    for (int t = 0; t < TT; t++) {
        size_t idx = base + (size_t)t * CC;
        if (tid < HSZ) {
            sr[tid] = r[idx + tid];  sw[tid] = dec[idx + tid];
            sk[tid] = k[idx + tid];  sv[tid] = v[idx + tid];
            sa[tid] = a[idx + tid];  sb[tid] = b[idx + tid];
        }
        __syncthreads();
        float sap = 0.0f;
#pragma unroll
        for (int j = 0; j < 16; j++) sap += S[j] * sa[kb + j];
        sap += __shfl_xor(sap, 1);
        sap += __shfl_xor(sap, 2);
        const float vt = sv[vi];
        float yp = 0.0f;
#pragma unroll
        for (int j = 0; j < 16; j++) {
            float s = S[j] * sw[kb + j] + sap * sb[kb + j] + vt * sk[kb + j];
            S[j] = s;
            yp += s * sr[kb + j];
        }
        yp += __shfl_xor(yp, 1);
        yp += __shfl_xor(yp, 2);
        if (kg == 0) y[idx + vi] = yp;
        __syncthreads();
    }
}

// =================================================================
// GroupNorm(64e-5) per head + rk*v bonus + gate -> bf16 (Wo input)
// =================================================================
__global__ __launch_bounds__(64) void k_post(const float* __restrict__ y,
        const float* __restrict__ r, const float* __restrict__ k2,
        const float* __restrict__ v, const float* __restrict__ g,
        const float* __restrict__ gnw, const float* __restrict__ gnb,
        const float* __restrict__ rkp, bf16* __restrict__ yg) {
    __shared__ float s0[64], s1[64], s2[64];
    int blk = blockIdx.x, i = threadIdx.x;
    size_t base = (size_t)blk * HSZ;
    int c = (blk % NH) * HSZ + i;
    float yv = y[base + i];
    float rkv = r[base + i] * k2[base + i] * rkp[c];
    s0[i] = yv; s1[i] = yv * yv; s2[i] = rkv;
    __syncthreads();
    for (int st = 32; st > 0; st >>= 1) {
        if (i < st) { s0[i] += s0[i+st]; s1[i] += s1[i+st]; s2[i] += s2[i+st]; }
        __syncthreads();
    }
    float mean = s0[0] * (1.0f / HSZ);
    float var  = s1[0] * (1.0f / HSZ) - mean * mean;
    float gn   = (yv - mean) * rsqrtf(var + 64e-5f) * gnw[c] + gnb[c];
    float o    = gn + s2[0] * v[base + i];
    yg[base + i] = (bf16)(o * g[base + i]);
}

// =================================================================
// WMMA GEMM:  C[M,N] = A[M,K] * W[N,K]^T  (+bias) (act) (+resid)
// bf16 operands, fp32 accumulation via v_wmma_f32_16x16x32_bf16.
// Block = 256 thr (8 waves), tile 128x64; each wave 2x2 16x16 tiles.
// Double-buffered K-loop:
//   A tile:  per-lane GLOBAL_LOAD_ASYNC_TO_LDS_B128   (ASYNCcnt)
//   B tile:  one TDM tensor_load_to_lds per K-step     (TENSORcnt),
//            issued by wave 0; descriptor pads each 64B row by 32B
//            to reproduce the 48-element LDS row stride in hardware.
// =================================================================
#define MBLK 128
#define NBLK 64
#define KBLK 32
#define LDST 48                     // padded LDS row stride (elements)
#define ABUFE (MBLK*LDST)           // elems per A buffer
#define BBUFE (NBLK*LDST)           // elems per B buffer
#define SMEMB ((2*ABUFE + 2*BBUFE) * 2)   // dynamic LDS bytes
#define BBASEB (2*ABUFE*2)          // byte offset of B buffers in LDS

__device__ __forceinline__ void async_ld16(unsigned lds_byte_off, const bf16* g) {
    // per-lane: LDS[lds_byte_off .. +15] = MEM[g .. g+15]
    asm volatile("global_load_async_to_lds_b128 %0, %1, off"
                 :: "v"(lds_byte_off), "v"(g) : "memory");
}
#define WAIT_ASYNC(n) asm volatile("s_wait_asynccnt " #n ::: "memory")

// TDM: load one 64-row x 32-elem bf16 tile of W (row stride K elems) into
// LDS at lds_byte, padding +32B after each 64B row (48-elem LDS stride).
// 2-group descriptor form => VADDR2/VADDR3 = NULL => 2D tensor.
__device__ __forceinline__ void tdm_load_b_tile(const bf16* gtile, unsigned lds_byte,
                                                int K, int Nrows) {
    unsigned long long ga = (unsigned long long)(uintptr_t)gtile;
    v4u g0;
    g0[0] = 1u;                                    // count=1 (valid), user mode
    g0[1] = lds_byte;                              // lds_addr (bytes)
    g0[2] = (unsigned)ga;                          // global_addr[31:0]
    g0[3] = ((unsigned)(ga >> 32) & 0x01FFFFFFu)   // global_addr[56:32]
          | (2u << 30);                            // type = 2 ("image")
    v8u g1;
    g1[0] = (1u << 16)      // data_size = 2 bytes
          | (1u << 20)      // pad_enable
          | (3u << 22)      // pad_interval: 16 DWORDs (one 64B tile row)
          | (7u << 25);     // pad_amount:   8 DWORDs (32B) -> 48-elem stride
    g1[1] = ((unsigned)K & 0xFFFFu) << 16;              // tensor_dim0[15:0]
    g1[2] = (((unsigned)K >> 16) & 0xFFFFu)             // tensor_dim0[31:16]
          | (((unsigned)Nrows & 0xFFFFu) << 16);        // tensor_dim1[15:0]
    g1[3] = (((unsigned)Nrows >> 16) & 0xFFFFu)         // tensor_dim1[31:16]
          | (32u << 16);                                // tile_dim0 = 32
    g1[4] = 64u;                                        // tile_dim1 = 64 rows
    g1[5] = (unsigned)K;                                // tensor_dim0_stride[31:0]
    g1[6] = 0u;
    g1[7] = 0u;
    asm volatile("tensor_load_to_lds %0, %1" :: "s"(g0), "s"(g1) : "memory");
}

__global__ __launch_bounds__(256) void k_gemm_bf16(
        const bf16* __restrict__ A, const bf16* __restrict__ W,
        const float* __restrict__ bias, const float* __restrict__ resid,
        float* __restrict__ Cout, bf16* __restrict__ Cbf,
        int M, int N, int K, int act) {
    extern __shared__ __align__(16) bf16 smem[];   // [2][ABUFE] A | [2][BBUFE] B
    bf16* sA = smem;
    bf16* sB = smem + 2 * ABUFE;

    const int tid  = threadIdx.x;
    const int lane = tid & 31;
    const int wave = tid >> 5;
    const int wm = wave & 3;       // 4 waves down M (32 rows each)
    const int wn = wave >> 2;      // 2 waves across N (32 cols each)
    const int m0 = blockIdx.y * MBLK;
    const int n0 = blockIdx.x * NBLK;
    const bool w0 = (wave == 0);   // TDM issuing wave

    v8f acc[2][2];
#pragma unroll
    for (int i = 0; i < 2; i++)
#pragma unroll
        for (int j = 0; j < 2; j++)
#pragma unroll
            for (int e = 0; e < 8; e++) acc[i][j][e] = 0.0f;

    // A staging: each thread moves one 16-elem K segment of one row
    const int ar   = tid >> 1;            // A row within tile (0..127)
    const int aseg = (tid & 1) * 16;      // K segment
    const bf16* gA = A + (size_t)(m0 + ar) * K + aseg;
    const unsigned ldsA = (unsigned)((ar * LDST + aseg) * 2);
    const bf16* gBtile = W + (size_t)n0 * K;   // B panel origin

    const int ntk = K / KBLK;
    auto issueA = [&](int buf, int k0) {
        unsigned la = ldsA + (unsigned)(buf * ABUFE * 2);
        async_ld16(la,      gA + k0);
        async_ld16(la + 16, gA + k0 + 8);
    };

    // prologue: tile 0 in flight (A via async lanes, B via TDM)
    issueA(0, 0);
    if (w0) tdm_load_b_tile(gBtile, BBASEB, K, N);

    const int hi16 = lane >> 4;       // lane half
    const int lr   = lane & 15;
    const int kbA  = hi16 * 16;       // A frag K base (contiguous, matches B)
    const int kbB  = hi16 * 16;       // B frag K base (contiguous, matches A)

    for (int it = 0; it < ntk; ++it) {
        const int cur = it & 1;
        if (it + 1 < ntk) {
            issueA(cur ^ 1, (it + 1) * KBLK);
            if (w0) tdm_load_b_tile(gBtile + (size_t)(it + 1) * KBLK,
                                    (unsigned)(BBASEB + (cur ^ 1) * BBUFE * 2), K, N);
            if (it + 2 < ntk)                          // warm L2 for tile+2
                __builtin_prefetch(gA + (size_t)(it + 2) * KBLK, 0, 1);
            // async loads retire in order: waiting down to the newly issued
            // count (2) guarantees tile `it`'s A data landed; wave0 likewise
            // allows 1 outstanding TDM op (the tile it+1 descriptor).
            WAIT_ASYNC(2);
            if (w0) __builtin_amdgcn_s_wait_tensorcnt(1);
        } else {
            WAIT_ASYNC(0);
            if (w0) __builtin_amdgcn_s_wait_tensorcnt(0);
        }
        __syncthreads();

        const bf16* cA = sA + cur * ABUFE;
        const bf16* cB = sB + cur * BBUFE;
        v16bf afr[2], bfr[2];
#pragma unroll
        for (int ti = 0; ti < 2; ti++) {
            const bf16* pa = &cA[(wm * 32 + ti * 16 + lr) * LDST + kbA];
#pragma unroll
            for (int e = 0; e < 8; e++) { afr[ti][e] = pa[e]; afr[ti][e + 8] = pa[e + 8]; }
        }
#pragma unroll
        for (int tj = 0; tj < 2; tj++) {
            const bf16* pb = &cB[(wn * 32 + tj * 16 + lr) * LDST + kbB];
#pragma unroll
            for (int e = 0; e < 16; e++) bfr[tj][e] = pb[e];
        }
#pragma unroll
        for (int ti = 0; ti < 2; ti++)
#pragma unroll
            for (int tj = 0; tj < 2; tj++)
                acc[ti][tj] = __builtin_amdgcn_wmma_f32_16x16x32_bf16(
                    false, afr[ti], false, bfr[tj], (short)0, acc[ti][tj], false, false);
        __syncthreads();
    }

    // ---- epilogue: C/D layout lane(N)=lane&15, M = vgpr + 8*(lane>=16) ----
#pragma unroll
    for (int ti = 0; ti < 2; ti++)
#pragma unroll
        for (int tj = 0; tj < 2; tj++) {
            const int n = n0 + wn * 32 + tj * 16 + lr;
#pragma unroll
            for (int e = 0; e < 8; e++) {
                const int m = m0 + wm * 32 + ti * 16 + e + hi16 * 8;
                float val = acc[ti][tj][e];
                if (bias) val += bias[n];
                if (act == 1) { val = fmaxf(val, 0.0f); val = val * val; }   // ReLU^2
                if (resid) val += resid[(size_t)m * N + n];
                if (Cout) Cout[(size_t)m * N + n] = val;
                if (Cbf)  Cbf[(size_t)m * N + n] = (bf16)val;
            }
        }
}

// =================================================================
// host orchestration
// =================================================================
extern "C" void kernel_launch(void* const* d_in, const int* in_sizes, int n_in,
                              void* d_out, int out_size, void* d_ws, size_t ws_size,
                              hipStream_t stream) {
    (void)in_sizes; (void)n_in; (void)out_size; (void)ws_size;
    const int*   tokens = (const int*)d_in[0];
    const float* emb    = (const float*)d_in[1];
    auto F = [&](int i) { return (const float*)d_in[i]; };

    // ---- workspace allocator ----
    char* wsp = (char*)d_ws;
    size_t off = 0;
    auto alloc = [&](size_t bytes) -> void* {
        off = (off + 255) & ~(size_t)255;
        void* p = wsp + off;
        off += bytes;
        return p;
    };

    const size_t NBTC = (size_t)BT * CC;
    float* x      = (float*)alloc(NBTC * 4);
    float* xn     = (float*)alloc(NBTC * 4);
    float* vfirst = (float*)alloc(NBTC * 4);
    float* rbuf   = (float*)alloc(NBTC * 4);
    float* kbuf   = (float*)alloc(NBTC * 4);
    float* vbuf   = (float*)alloc(NBTC * 4);
    float* decb   = (float*)alloc(NBTC * 4);
    float* aab    = (float*)alloc(NBTC * 4);
    float* gbuf   = (float*)alloc(NBTC * 4);
    float* vgate  = (float*)alloc(NBTC * 4);
    float* k2b    = (float*)alloc(NBTC * 4);
    float* ainb   = (float*)alloc(NBTC * 4);
    float* binb   = (float*)alloc(NBTC * 4);
    float* ybuf   = (float*)alloc(NBTC * 4);
    float* hlora  = (float*)alloc((size_t)BT * LR * 4);

    bf16* xr  = (bf16*)alloc(NBTC * 2);
    bf16* xw  = (bf16*)alloc(NBTC * 2);
    bf16* xk  = (bf16*)alloc(NBTC * 2);
    bf16* xv  = (bf16*)alloc(NBTC * 2);
    bf16* xa  = (bf16*)alloc(NBTC * 2);
    bf16* xg  = (bf16*)alloc(NBTC * 2);
    bf16* yg  = (bf16*)alloc(NBTC * 2);
    bf16* xfn = (bf16*)alloc(NBTC * 2);
    bf16* xh  = (bf16*)alloc(NBTC * 2);
    bf16* hff = (bf16*)alloc((size_t)BT * FFD * 2);

    const size_t WCC = (size_t)CC * CC;
    bf16* tw[2][4];
    for (int l = 0; l < 2; l++) for (int j = 0; j < 4; j++) tw[l][j] = (bf16*)alloc(WCC * 2);
    bf16* fw[2][2];
    for (int l = 0; l < 2; l++) for (int j = 0; j < 2; j++) fw[l][j] = (bf16*)alloc((size_t)CC * FFD * 2);
    bf16* whead = (bf16*)alloc((size_t)VOC * CC * 2);

    // ---- convert weights fp32 -> bf16 (once per launch) ----
    auto cvt = [&](const float* s, bf16* dptr, size_t n) {
        unsigned g = (unsigned)((n + 1023) / 1024);
        k_cvt<<<dim3(g), 256, 0, stream>>>(s, dptr, n);
    };
    const int tbase[2] = {12, 38};   // tmix1 / tmix2 param base indices
    const int fbase[2] = {64, 67};   // ffn1 / ffn2
    for (int l = 0; l < 2; l++)
        for (int j = 0; j < 4; j++) cvt(F(tbase[l] + 20 + j), tw[l][j], WCC);   // Wr,Wk,Wv,Wo
    for (int l = 0; l < 2; l++) {
        cvt(F(fbase[l] + 1), fw[l][0], (size_t)FFD * CC);                        // ffn Wk
        cvt(F(fbase[l] + 2), fw[l][1], (size_t)CC * FFD);                        // ffn Wv
    }
    cvt(F(70), whead, (size_t)VOC * CC);

    auto gemm = [&](const bf16* A, const bf16* Wt, const float* bias, const float* resid,
                    float* Co, bf16* Cb, int M, int N, int K, int act) {
        k_gemm_bf16<<<dim3((unsigned)(N / NBLK), (unsigned)(M / MBLK)), 256, SMEMB, stream>>>(
            A, Wt, bias, resid, Co, Cb, M, N, K, act);
    };

    // ---- embedding ----
    k_embed<<<dim3(BT), 256, 0, stream>>>(tokens, emb, x);

    auto run_tmix = [&](int tp, int lnw, bf16* const* Wts, bool first) {
        k_layernorm<<<dim3(BT), 256, 0, stream>>>(x, F(lnw), F(lnw + 1), xn, (bf16*)nullptr);
        k_tmix_mix<<<dim3(16384), 256, 0, stream>>>(xn,
            F(tp + 0), F(tp + 1), F(tp + 2), F(tp + 3), F(tp + 4), F(tp + 5),
            xr, xw, xk, xv, xa, xg);
        gemm(xr, Wts[0], nullptr, nullptr, rbuf, nullptr, BT, CC, CC, 0);
        gemm(xk, Wts[1], nullptr, nullptr, kbuf, nullptr, BT, CC, CC, 0);
        gemm(xv, Wts[2], nullptr, nullptr, vbuf, nullptr, BT, CC, CC, 0);
        // decay lora: tanh(xw@w1)@w2 + w0 -> dec
        k_lora_s1<<<dim3(BT), 256, 0, stream>>>(xw, F(tp + 6), hlora, 1);
        k_lora_s2<<<dim3(16384), 256, 0, stream>>>(hlora, F(tp + 7), F(tp + 8), decb, 2);
        // a lora: sigmoid(a0 + (xa@a1)@a2)
        k_lora_s1<<<dim3(BT), 256, 0, stream>>>(xa, F(tp + 9), hlora, 0);
        k_lora_s2<<<dim3(16384), 256, 0, stream>>>(hlora, F(tp + 10), F(tp + 11), aab, 1);
        // g lora: sigmoid(xg@g1)@g2
        k_lora_s1<<<dim3(BT), 256, 0, stream>>>(xg, F(tp + 15), hlora, 2);
        k_lora_s2<<<dim3(16384), 256, 0, stream>>>(hlora, F(tp + 16), nullptr, gbuf, 0);
        if (first) {
            hipMemcpyAsync(vfirst, vbuf, NBTC * 4, hipMemcpyDeviceToDevice, stream);
        } else {
            k_lora_s1<<<dim3(BT), 256, 0, stream>>>(xv, F(tp + 12), hlora, 0);
            k_lora_s2<<<dim3(16384), 256, 0, stream>>>(hlora, F(tp + 13), F(tp + 14), vgate, 1);
            k_vmix<<<dim3(16384), 256, 0, stream>>>(vbuf, vfirst, vgate);
        }
        k_kv_prep<<<dim3(BT * NH), 64, 0, stream>>>(kbuf, aab, F(tp + 17), F(tp + 18),
                                                    k2b, ainb, binb);
        k_rwkv_scan<<<dim3(BB * NH), 256, 0, stream>>>(rbuf, decb, k2b, vbuf, ainb, binb, ybuf);
        k_post<<<dim3(BT * NH), 64, 0, stream>>>(ybuf, rbuf, k2b, vbuf, gbuf,
                                                 F(tp + 24), F(tp + 25), F(tp + 19), yg);
        gemm(yg, Wts[3], nullptr, x, x, nullptr, BT, CC, CC, 0);   // x += (y*g)@Wo^T
    };

    auto run_ffn = [&](int fp, int lnw, bf16* const* Wts) {
        k_layernorm<<<dim3(BT), 256, 0, stream>>>(x, F(lnw), F(lnw + 1), xn, (bf16*)nullptr);
        k_ffn_mix<<<dim3(16384), 256, 0, stream>>>(xn, F(fp), xfn);
        gemm(xfn, Wts[0], nullptr, nullptr, nullptr, hff, BT, FFD, CC, 1);   // ReLU^2 -> bf16
        gemm(hff, Wts[1], nullptr, x, x, nullptr, BT, CC, FFD, 0);           // x += h@Wv^T
    };

    run_tmix(12, 2, tw[0], true);
    run_ffn(64, 4, fw[0]);
    run_tmix(38, 6, tw[1], false);
    run_ffn(67, 8, fw[1]);

    // ---- final LN + head ----
    k_layernorm<<<dim3(BT), 256, 0, stream>>>(x, F(10), F(11), xn, xh);
    gemm(xh, whead, F(71), nullptr, (float*)d_out, nullptr, BT, VOC, CC, 0);
}
